// W4A8B16O16Linear_40243843564264
// MI455X (gfx1250) — 8-bit matrix operand order repaired and checked by reading, compiled, not run on hardware
//
#include <hip/hip_runtime.h>


typedef int v8i __attribute__((ext_vector_type(8)));

#define GROUP    128
#define MDIM     64
#define KDIM     4096
#define NDIM     11008
#define NGROUPS  (KDIM / GROUP)
#define BLOCK_N  128
#define LDS_ROWSTRIDE 33
#define KSPLIT   4
#define GPC      (NGROUPS / KSPLIT)
#define MN       (MDIM * NDIM)

template <bool SPLIT>
__global__ __launch_bounds__(256)
void w4a8_gemm_kernel(const int* __restrict__ x,
                      const int* __restrict__ qw,
                      const float* __restrict__ input_scale_p,
                      const float* __restrict__ s_channel,
                      const _Float16* __restrict__ s_group,
                      _Float16* __restrict__ out,
                      float* __restrict__ partial)
{
    __shared__ int lds_a[MDIM * LDS_ROWSTRIDE];

    const int tid  = threadIdx.x;
    const int wave = tid >> 5;
    const int lane = tid & 31;
    const int half = lane >> 4;
    const int l16  = lane & 15;

    const int n = blockIdx.x * BLOCK_N + wave * 16 + l16;

    const int g0 = SPLIT ? (int)blockIdx.y * GPC : 0;
    const int g1 = SPLIT ? g0 + GPC : NGROUPS;

    float facc[4][8];
    #pragma unroll
    for (int mt = 0; mt < 4; ++mt)
        #pragma unroll
        for (int i = 0; i < 8; ++i) facc[mt][i] = 0.0f;

    for (int g = g0; g < g1; ++g) {
        __syncthreads();
        #pragma unroll
        for (int it = 0; it < 8; ++it) {
            const int idx = it * 256 + tid;
            const int row = idx >> 5;
            const int dw  = idx & 31;
            const int4 a4 = *reinterpret_cast<const int4*>(
                x + (size_t)row * KDIM + g * GROUP + dw * 4);
            lds_a[row * LDS_ROWSTRIDE + dw] =
                ( a4.x        & 0xFF)        | ((a4.y & 0xFF) << 8) |
                ((a4.z & 0xFF) << 16)        | ((a4.w & 0xFF) << 24);
        }
        __syncthreads();

        if (g + 1 < g1) {
            __builtin_prefetch(qw + (size_t)((g + 1) * GROUP) * NDIM + n, 0, 1);
        }

        v8i bfrag[2];
        #pragma unroll
        for (int h = 0; h < 2; ++h) {
            #pragma unroll
            for (int v = 0; v < 8; ++v) {
                const int kb = g * GROUP + h * 64 +
                               (v & 1) * 4 + ((v >> 1) << 4) + half * 8;
                const int* qp = qw + (size_t)kb * NDIM + n;
                const int q0 = qp[0];
                const int q1 = qp[(size_t)NDIM];
                const int q2 = qp[(size_t)2 * NDIM];
                const int q3 = qp[(size_t)3 * NDIM];
                bfrag[h][v] = (((q0 - 8) & 0xFF))        |
                              (((q1 - 8) & 0xFF) << 8)   |
                              (((q2 - 8) & 0xFF) << 16)  |
                              (((q3 - 8) & 0xFF) << 24);
            }
        }

        const float sg = (float)s_group[(size_t)g * NDIM + n];

        #pragma unroll
        for (int mt = 0; mt < 4; ++mt) {
            v8i iacc = {};
            #pragma unroll
            for (int h = 0; h < 2; ++h) {
                v8i afrag;
                const int rowbase = (mt * 16 + l16) * LDS_ROWSTRIDE;
                const int dwb     = h * 16 + half * 2;
                #pragma unroll
                for (int p = 0; p < 4; ++p) {
                    const int d0 = rowbase + dwb + p * 4;
                    afrag[2 * p]     = lds_a[d0];
                    afrag[2 * p + 1] = lds_a[d0 + 1];
                }
                iacc = __builtin_amdgcn_wmma_i32_16x16x64_iu8(
                    true, afrag, true, bfrag[h], iacc, false, false);
            }
            #pragma unroll
            for (int i = 0; i < 8; ++i)
                facc[mt][i] += (float)iacc[i] * sg;
        }
    }

    if (SPLIT) {
        float* dst = partial + (size_t)blockIdx.y * MN;
        #pragma unroll
        for (int mt = 0; mt < 4; ++mt) {
            #pragma unroll
            for (int i = 0; i < 8; ++i) {
                const int m = mt * 16 + half * 8 + i;
                dst[(size_t)m * NDIM + n] = facc[mt][i];
            }
        }
    } else {
        const float scale = input_scale_p[0] * s_channel[n];
        #pragma unroll
        for (int mt = 0; mt < 4; ++mt) {
            #pragma unroll
            for (int i = 0; i < 8; ++i) {
                const int m = mt * 16 + half * 8 + i;
                out[(size_t)m * NDIM + n] = (_Float16)(facc[mt][i] * scale);
            }
        }
    }
}

__global__ __launch_bounds__(256)
void w4a8_reduce_kernel(const float* __restrict__ partial,
                        const float* __restrict__ input_scale_p,
                        const float* __restrict__ s_channel,
                        _Float16* __restrict__ out)
{
    const int id = blockIdx.x * 256 + threadIdx.x;
    if (id >= MN) return;
    const int n = id % NDIM;
    float s = 0.0f;
    #pragma unroll
    for (int k = 0; k < KSPLIT; ++k)
        s += partial[(size_t)k * MN + id];
    out[id] = (_Float16)(s * input_scale_p[0] * s_channel[n]);
}

extern "C" void kernel_launch(void* const* d_in, const int* in_sizes, int n_in,
                              void* d_out, int out_size, void* d_ws, size_t ws_size,
                              hipStream_t stream) {
    const int*      x   = (const int*)d_in[0];
    const int*      qw  = (const int*)d_in[1];
    const float*    isc = (const float*)d_in[2];
    const float*    sch = (const float*)d_in[3];
    const _Float16* sg  = (const _Float16*)d_in[4];
    _Float16*       out = (_Float16*)d_out;
    float*          ws  = (float*)d_ws;

    const size_t need = (size_t)KSPLIT * MN * sizeof(float);

    if (ws_size >= need) {
        dim3 grid(NDIM / BLOCK_N, KSPLIT);
        dim3 block(256);
        hipLaunchKernelGGL((w4a8_gemm_kernel<true>), grid, block, 0, stream,
                           x, qw, isc, sch, sg, out, ws);
        dim3 rgrid((MN + 255) / 256);
        hipLaunchKernelGGL(w4a8_reduce_kernel, rgrid, block, 0, stream,
                           ws, isc, sch, out);
    } else {
        dim3 grid(NDIM / BLOCK_N);
        dim3 block(256);
        hipLaunchKernelGGL((w4a8_gemm_kernel<false>), grid, block, 0, stream,
                           x, qw, isc, sch, sg, out, ws);
    }
}
